// LookupConv2d_22952305230322
// MI455X (gfx1250) — matrix operand order repaired and checked by reading, not compiled, not run on hardware
//
#include <hip/hip_runtime.h>

typedef __attribute__((ext_vector_type(16))) _Float16 v16h;
typedef __attribute__((ext_vector_type(8)))  float    v8f;
typedef __attribute__((ext_vector_type(8)))  _Float16 v8h;

#define B_  32
#define CIN 64
#define HH  64
#define WW  64
#define DD  256
#define OO  512
#define SSP 4

// ---------------------------------------------------------------------------
// Kernel 1: fold the sparse lookup into effective filters
//   W_eff[o,c,kh,kw] = sum_s coeff[o,s] * dict[idx[o,s], c, kh, kw]
// stored as f16 pre-swizzled WMMA A-fragments:
//   Apack[((mt*18 + kstep)*32 + lane)*16 + j]
// (lane, j) follow the CDNA5 16-bit A-matrix 16x32 VGPR layout:
//   lanes 0-15 : K = {0..7, 16..23},  lanes 16-31 : K = {8..15, 24..31}
// ---------------------------------------------------------------------------
__global__ __launch_bounds__(256)
void pack_weights_kernel(const float* __restrict__ dict,
                         const float* __restrict__ coeff,
                         const int*   __restrict__ idx,
                         _Float16*    __restrict__ Apack) {
    int t = blockIdx.x * blockDim.x + threadIdx.x;   // 0 .. 32*18*32*16-1
    int j     = t & 15;
    int lane  = (t >> 4) & 31;
    int kstep = (t >> 9) % 18;
    int mt    = t / (16 * 32 * 18);

    int khalf = lane >> 4;
    int K32   = ((j < 8) ? j : j + 8) + khalf * 8;   // ISA A-fragment K index
    int tap   = kstep >> 1;
    int chalf = kstep & 1;
    int c  = chalf * 32 + K32;
    int o  = mt * 16 + (lane & 15);
    int kh = tap / 3, kw = tap % 3;

    float acc = 0.f;
#pragma unroll
    for (int s = 0; s < SSP; ++s) {
        int d = idx[o * SSP + s];
        acc += coeff[o * SSP + s] * dict[((d * CIN + c) * 3 + kh) * 3 + kw];
    }
    Apack[t] = (_Float16)acc;
}

// ---------------------------------------------------------------------------
// Kernel 2: implicit-GEMM 3x3 conv via v_wmma_f32_16x16x32_f16.
// Block = 256 threads (8 waves). blockIdx.x selects a 256-channel M block,
// blockIdx.y = (b,h) row. Each wave owns TWO 16-row M tiles x four 16-pixel
// N tiles (8 f32 accumulator tiles); the 4 B-fragments per K step feed 8
// WMMAs. A (global) and B (LDS) fragments are double-buffered one K step
// ahead. All addresses are ONE per-lane base pointer per stream + literal
// offsets, so address math folds into the instruction offset fields.
// ---------------------------------------------------------------------------
__global__ __launch_bounds__(256)
void lookup_conv_kernel(const float*    __restrict__ x,
                        const _Float16* __restrict__ Apack,
                        const float*    __restrict__ bias,
                        float*          __restrict__ out) {
    __shared__ alignas(32) _Float16 xs[3 * 66 * CIN];   // [r][wcol 0..65][c]

    const int mblk = blockIdx.x;        // 0..1  -> 256 output channels
    const int bh   = blockIdx.y;        // 0..B*H-1
    const int b = bh >> 6;
    const int h = bh & 63;

    const int tid  = threadIdx.x;
    const int wave = tid >> 5;
    const int lane = tid & 31;

    // ---- stage x[b, :, h-1..h+1, :] into LDS, transposed, zero-padded ----
    if (tid < 192) {
        const int c  = tid & 63;
        const int r  = tid >> 6;        // 0..2
        const int hr = h + r - 1;
        _Float16* row = &xs[(r * 66) * CIN + c];
        row[0]        = (_Float16)0.f;  // wcol 0  (left halo)
        row[65 * CIN] = (_Float16)0.f;  // wcol 65 (right halo)
        if (hr >= 0 && hr < HH) {
            const float4* src = (const float4*)&x[((b * CIN + c) * HH + hr) * WW];
#pragma unroll 4
            for (int i = 0; i < 16; ++i) {
                float4 v = src[i];
                row[(i * 4 + 1) * CIN] = (_Float16)v.x;
                row[(i * 4 + 2) * CIN] = (_Float16)v.y;
                row[(i * 4 + 3) * CIN] = (_Float16)v.z;
                row[(i * 4 + 4) * CIN] = (_Float16)v.w;
            }
        } else {
#pragma unroll 8
            for (int w = 1; w <= 64; ++w) row[w * CIN] = (_Float16)0.f;
        }
    }
    __syncthreads();

    const int mt0   = mblk * 16 + wave * 2;  // first of two M tiles (0..31)
    const int nlo   = lane & 15;             // pixel within N tile
    const int khalf = lane >> 4;             // ISA B-fragment K-half select

    // Single per-lane base pointers; everything else is a literal offset.
    const _Float16* ApBase =
        Apack + ((size_t)(mt0 * 18) * 32 + lane) * 16;   // +mi*9216 +ks*512
    const _Float16* xsBase = &xs[nlo * CIN + khalf * 8];

    v8f  acc[2][4] = {};
    v16h abuf[2][2];                         // [parity][mi]
    v16h bbuf[2][4];                         // [parity][nt]

    auto loadA = [&](int ks, v16h* a) {
        a[0] = *(const v16h*)(ApBase +        ks * 512);
        a[1] = *(const v16h*)(ApBase + 9216 + ks * 512);
    };
    auto loadB = [&](int ks, v16h* bf) {
        const int tap = ks >> 1, chalf = ks & 1;
        const int kh = tap / 3, kw = tap % 3;
        const int off = (kh * 66 + kw) * CIN + chalf * 32;  // literal per ks
        auto ld = [](const _Float16* p) -> v16h {
            v8h lo = *(const v8h*)p, hi = *(const v8h*)(p + 16);
            return __builtin_shufflevector(lo, hi, 0,1,2,3,4,5,6,7,8,9,10,11,12,13,14,15);
        };
        bf[0] = ld(xsBase + off +  0 * CIN);
        bf[1] = ld(xsBase + off + 16 * CIN);
        bf[2] = ld(xsBase + off + 32 * CIN);
        bf[3] = ld(xsBase + off + 48 * CIN);
    };

    loadA(0, abuf[0]);
    loadB(0, bbuf[0]);

#pragma unroll
    for (int ks = 0; ks < 18; ++ks) {
        const int cur = ks & 1, nxt = cur ^ 1;
        if (ks < 17) {                      // prefetch next K step
            loadA(ks + 1, abuf[nxt]);
            loadB(ks + 1, bbuf[nxt]);
        }
#pragma unroll
        for (int mi = 0; mi < 2; ++mi)
#pragma unroll
            for (int nt = 0; nt < 4; ++nt)
                acc[mi][nt] = __builtin_amdgcn_wmma_f32_16x16x32_f16(
                    false, abuf[cur][mi], false, bbuf[cur][nt],
                    (short)0, acc[mi][nt], false, false);
    }

    // ---- epilogue: bias + store (C/D layout: M = v + 8*khalf, N = lane%16) ----
    // One base pointer each; mi/v/nt become literal offsets.
    const int   obase0 = mt0 * 16 + khalf * 8;
    const float* bb    = bias + obase0;                       // +mi*16 +v
    float*       ob    =
        out + (((size_t)b * OO + obase0) * HH + h) * WW + nlo; // +(mi*16+v)*4096 +nt*16

#pragma unroll
    for (int mi = 0; mi < 2; ++mi)
#pragma unroll
        for (int v = 0; v < 8; ++v) {
            const float bz = bb[mi * 16 + v];
            float* op = ob + (mi * 16 + v) * (HH * WW);
            op[ 0] = acc[mi][0][v] + bz;
            op[16] = acc[mi][1][v] + bz;
            op[32] = acc[mi][2][v] + bz;
            op[48] = acc[mi][3][v] + bz;
        }
}

// ---------------------------------------------------------------------------
extern "C" void kernel_launch(void* const* d_in, const int* in_sizes, int n_in,
                              void* d_out, int out_size, void* d_ws, size_t ws_size,
                              hipStream_t stream) {
    const float* x     = (const float*)d_in[0];   // [32,64,64,64]
    const float* dict  = (const float*)d_in[1];   // [256,64,3,3]
    const float* coeff = (const float*)d_in[2];   // [512,4]
    const float* bias  = (const float*)d_in[3];   // [512]
    const int*   idx   = (const int*)d_in[4];     // [512,4]
    float*       out   = (float*)d_out;           // [32,512,64,64]

    _Float16* Apack = (_Float16*)d_ws;            // 512*576 f16 = 576 KB

    // 32 m-tiles * 18 ksteps * 32 lanes * 16 halves = 294912 elements
    pack_weights_kernel<<<294912 / 256, 256, 0, stream>>>(dict, coeff, idx, Apack);

    dim3 grid(OO / 256, B_ * HH);                 // (2, 2048)
    lookup_conv_kernel<<<grid, 256, 0, stream>>>(x, Apack, bias, out);
}
